// CustomAttention_76527727280562
// MI455X (gfx1250) — matrix operand order repaired and checked by reading, not compiled, not run on hardware
//
#include <hip/hip_runtime.h>

// MI455X / gfx1250, wave32.
//   K0: fp32 -> bf16 conversion of hs and Wq/Wk/Wv (done once; weights are
//       reused 512x by the GEMM, so convert outside the hot loop)
//   K1: QKV projection GEMM, bf16 WMMA + f32 acc, 16x128 tile per wave,
//       fused bias + scatter to attention layouts (Q,K: [bh][s][d], V: [bh][d][s])
//   K2: flash attention: 4 waves/WG share one (b,head); K/V chunks staged in
//       LDS by the Tensor Data Mover (tensor_load_to_lds + s_wait_tensorcnt),
//       double-buffered; online softmax; P transposed through per-wave LDS.

typedef __attribute__((ext_vector_type(16))) __bf16 v16bf;
typedef __attribute__((ext_vector_type(8)))  __bf16 v8bf;
typedef __attribute__((ext_vector_type(8)))  float  v8f;
typedef __attribute__((ext_vector_type(4)))  float  v4f;
typedef __attribute__((ext_vector_type(4)))  unsigned int u32x4;
typedef __attribute__((ext_vector_type(8)))  int i32x8;
typedef __attribute__((ext_vector_type(4)))  int i32x4;

#define B_  8
#define S_  1024
#define H_  1024
#define NH_ 16
#define HD_ 64

union AFrag { v16bf v; v8bf h[2]; __bf16 e[16]; };
union BFrag { v16bf v; v8bf h[2]; __bf16 e[16]; };
union CFrag { v8f   v; float f[8]; };

// ---------------------------------------------------------------------------
// TDM 2D tile load: global (bf16, row stride stride0 elements) -> LDS.
// D# per CDNA5 ISA ch.8: group0 = {flags, lds_addr, global_addr, type=2},
// group1 = {data_size=2B, tensor dims, tile dims, dim0 stride}. Tensor dims
// are set to the tile dims (tiles are always fully in-bounds here).
// ---------------------------------------------------------------------------
__device__ __forceinline__ void tdm_load_2d(unsigned int lds_off, const void* gptr,
                                            unsigned int tile0, unsigned int tile1,
                                            unsigned long long stride0)
{
    unsigned long long ga = (unsigned long long)gptr;
    u32x4 g0;
    g0[0] = 1u;                                   // count=1, no gather
    g0[1] = lds_off;                              // LDS byte address
    g0[2] = (unsigned int)ga;                     // global_addr[31:0]
    g0[3] = (unsigned int)((ga >> 32) & 0x1FFFFFFu) | 0x80000000u; // [56:32], type=2
    i32x8 g1;
    g1[0] = 0x00010000;                           // data_size = 2 bytes
    g1[1] = (int)((tile0 & 0xFFFFu) << 16);       // tensor_dim0[15:0]
    g1[2] = (int)(((tile0 >> 16) & 0xFFFFu) | ((tile1 & 0xFFFFu) << 16)); // dim0 hi | dim1 lo
    g1[3] = (int)(((tile1 >> 16) & 0xFFFFu) | (tile0 << 16));             // dim1 hi | tile_dim0
    g1[4] = (int)tile1;                           // tile_dim1 (tile_dim2 = 0)
    g1[5] = (int)(unsigned int)(stride0 & 0xFFFFFFFFu);    // dim0 stride lo
    g1[6] = (int)(unsigned int)((stride0 >> 32) & 0xFFFFu);// dim0 stride hi
    g1[7] = 0;
    i32x4 z4 = {0, 0, 0, 0};
#if __clang_major__ >= 23
    i32x8 z8 = {0, 0, 0, 0, 0, 0, 0, 0};
    __builtin_amdgcn_tensor_load_to_lds(g0, g1, z4, z4, z8, 0);
#else
    __builtin_amdgcn_tensor_load_to_lds(g0, g1, z4, z4, 0);
#endif
}

// ---------------------------------------------------------------------------
// Kernel 0: fp32 -> bf16, 8 elements/thread.
// ---------------------------------------------------------------------------
__global__ __launch_bounds__(256) void cvt_bf16_kernel(
    const float* __restrict__ src, __bf16* __restrict__ dst, int n8)
{
    const int i = blockIdx.x * 256 + threadIdx.x;
    if (i >= n8) return;
    v4f a = *((const v4f*)src + 2 * i);
    v4f b = *((const v4f*)src + 2 * i + 1);
    v8bf o;
    #pragma unroll
    for (int j = 0; j < 4; ++j) { o[j] = (__bf16)a[j]; o[4 + j] = (__bf16)b[j]; }
    *((v8bf*)dst + i) = o;
}

// ---------------------------------------------------------------------------
// Kernel 1: Q/K/V = hs @ W^T + b (all bf16 in, f32 acc, bf16 out).
// One wave -> (matrix, 16-token row tile, 128-wide output strip): 8 WMMAs per
// K-step, one shared A fragment. K-loop of 32 steps.
// A frag element e -> K = 16*half + e                  (contiguous, same as B)
// B frag element e -> K = 16*half + e                  (contiguous, same as A)
// ---------------------------------------------------------------------------
__global__ __launch_bounds__(128) void qkv_proj_kernel(
    const __bf16* __restrict__ hsb,
    const __bf16* __restrict__ Wqb, const __bf16* __restrict__ Wkb,
    const __bf16* __restrict__ Wvb,
    const float* __restrict__ bq, const float* __restrict__ bk,
    const float* __restrict__ bv,
    __bf16* __restrict__ Qb, __bf16* __restrict__ Kb, __bf16* __restrict__ VTb)
{
    const int lane = threadIdx.x & 31;
    const int wave = threadIdx.x >> 5;
    const int w    = blockIdx.x * 4 + wave;      // 0 .. 3*512*8-1
    const int mat       = w / (512 * 8);
    const int rem       = w % (512 * 8);
    const int row_tile  = rem / 8;
    const int col_strip = rem % 8;               // 128-wide strip

    const int half = lane >> 4;
    const int ln   = lane & 15;

    const __bf16* W;  const float* bias;
    if (mat == 0)      { W = Wqb; bias = bq; }
    else if (mat == 1) { W = Wkb; bias = bk; }
    else               { W = Wvb; bias = bv; }

    const int row0 = row_tile * 16;
    const int col0 = col_strip * 128;

    CFrag acc[8];
    #pragma unroll
    for (int t = 0; t < 8; ++t)
        for (int r = 0; r < 8; ++r) acc[t].f[r] = 0.f;

    const __bf16* arow = hsb + (size_t)(row0 + ln) * H_;   // A row M = ln

    for (int kb = 0; kb < H_; kb += 32) {
        __builtin_prefetch(arow + kb + 32, 0, 0);          // global_prefetch_b8
        AFrag a;
        a.h[0] = *(const v8bf*)(arow + kb + 16 * half);
        a.h[1] = *(const v8bf*)(arow + kb + 16 * half + 8);
        #pragma unroll
        for (int t = 0; t < 8; ++t) {
            // B[k][n] = W[n][k]; lane column n=ln, o = col0 + t*16 + ln
            const __bf16* wrow = W + (size_t)(col0 + t * 16 + ln) * H_ + kb + 16 * half;
            BFrag bf;
            bf.h[0] = *(const v8bf*)(wrow);
            bf.h[1] = *(const v8bf*)(wrow + 8);
            acc[t].v = __builtin_amdgcn_wmma_f32_16x16x32_bf16(
                false, a.v, false, bf.v, (short)0, acc[t].v, false, false);
        }
    }

    // Epilogue: bias add, bf16 pack, scatter to attention layouts.
    #pragma unroll
    for (int t = 0; t < 8; ++t) {
        const int o    = col0 + t * 16 + ln;
        const int head = o >> 6;
        const int d    = o & 63;
        const float bval = bias[o];
        #pragma unroll
        for (int r = 0; r < 8; ++r) {
            const int token = row0 + r + 8 * half;   // C row = r + 8*half
            const int b = token >> 10;
            const int s = token & 1023;
            const __bf16 q = (__bf16)(acc[t].f[r] + bval);
            const size_t bh = (size_t)b * NH_ + head;
            if (mat == 2)      VTb[(bh * HD_ + d) * S_ + s] = q;   // V^T
            else if (mat == 1) Kb[(bh * S_ + s) * HD_ + d]  = q;
            else               Qb[(bh * S_ + s) * HD_ + d]  = q;
        }
    }
}

// ---------------------------------------------------------------------------
// Kernel 2: flash attention. 4 waves/WG share one (b,head); wave i owns rows
// [rg*64 + i*16, +16). Per 32-column KV chunk: TDM stages K (32x64) and V^T
// (64x32) into LDS (double-buffered, s_wait_tensorcnt + workgroup barrier),
// waves do S = Q K^T (4 wmma), online softmax, P via per-wave LDS transpose
// (s_wait_dscnt), O += P V (4 wmma). Final O /= rowsum.
// ---------------------------------------------------------------------------
__global__ __launch_bounds__(128) void flash_attn_kernel(
    const __bf16* __restrict__ Qb, const __bf16* __restrict__ Kb,
    const __bf16* __restrict__ VTb, float* __restrict__ out)
{
    __shared__ __align__(16) __bf16 kbuf[2][32 * 64];   // [s_local][d]
    __shared__ __align__(16) __bf16 vbuf[2][64 * 32];   // [d][s_local]
    __shared__ __align__(16) __bf16 pb[4][16 * 32];     // per-wave P scratch

    const int lane = threadIdx.x & 31;
    const int wave = threadIdx.x >> 5;
    const int wg   = blockIdx.x;              // 0 .. 2047
    const int bh   = wg >> 4;                 // b*16 + head
    const int rg   = wg & 15;                 // 64-row group
    const int s0   = rg * 64 + wave * 16;

    const int half = lane >> 4;
    const int ln   = lane & 15;

    const __bf16* qbase  = Qb  + ((size_t)bh * S_ + s0) * HD_;
    const __bf16* kbase  = Kb  + (size_t)bh * S_ * HD_;
    const __bf16* vtbase = VTb + (size_t)bh * HD_ * S_;

    // LDS byte offsets for TDM (generic LDS addr truncates to LDS offset)
    const unsigned int koff[2] = { (unsigned int)(size_t)&kbuf[0][0],
                                   (unsigned int)(size_t)&kbuf[1][0] };
    const unsigned int voff[2] = { (unsigned int)(size_t)&vbuf[0][0],
                                   (unsigned int)(size_t)&vbuf[1][0] };

    // Q fragments, A layout, rows M = ln, K = head-dim (2 frags of K=32)
    AFrag qa[2];
    {
        const __bf16* qrow = qbase + (size_t)ln * HD_;
        #pragma unroll
        for (int kk = 0; kk < 2; ++kk) {
            qa[kk].h[0] = *(const v8bf*)(qrow + kk * 32 + 16 * half);
            qa[kk].h[1] = *(const v8bf*)(qrow + kk * 32 + 16 * half + 8);
        }
    }

    CFrag o[4];
    #pragma unroll
    for (int f = 0; f < 4; ++f)
        for (int r = 0; r < 8; ++r) o[f].f[r] = 0.f;
    float m[8], l[8];
    #pragma unroll
    for (int r = 0; r < 8; ++r) { m[r] = -1e30f; l[r] = 0.f; }

    __bf16* pbuf = pb[wave];
    const float scale = 0.125f;   // 1/sqrt(64)

    // Prologue: stage chunk 0
    if (wave == 0) {
        tdm_load_2d(koff[0], kbase, 64, 32, 64);           // K rows: 32 x (64 d)
        tdm_load_2d(voff[0], vtbase, 32, 64, S_);          // V^T rows: 64 x (32 s)
        __builtin_amdgcn_s_wait_tensorcnt(0);
    }
    __syncthreads();

    int cur = 0;
    for (int c0 = 0; c0 < S_; c0 += 32) {
        const int nxt = cur ^ 1;
        if (wave == 0 && (c0 + 32) < S_) {                 // stage next chunk
            tdm_load_2d(koff[nxt], kbase + (size_t)(c0 + 32) * HD_, 64, 32, 64);
            tdm_load_2d(voff[nxt], vtbase + (c0 + 32), 32, 64, S_);
        }

        // ---- scores: 16x64 @ 64x32 -> two 16x16 C tiles (K frags from LDS)
        CFrag sc[2];
        #pragma unroll
        for (int t = 0; t < 2; ++t)
            for (int r = 0; r < 8; ++r) sc[t].f[r] = 0.f;
        const __bf16* kl = &kbuf[cur][0];
        #pragma unroll
        for (int t = 0; t < 2; ++t) {
            const __bf16* krow = kl + (t * 16 + ln) * 64;
            #pragma unroll
            for (int kk = 0; kk < 2; ++kk) {
                BFrag kf;
                kf.h[0] = *(const v8bf*)(krow + kk * 32 + 16 * half);
                kf.h[1] = *(const v8bf*)(krow + kk * 32 + 16 * half + 8);
                sc[t].v = __builtin_amdgcn_wmma_f32_16x16x32_bf16(
                    false, qa[kk].v, false, kf.v, (short)0, sc[t].v, false, false);
            }
        }

        // ---- online softmax (rows live across 16 lanes) -------------------
        float p0[8], p1[8];
        #pragma unroll
        for (int r = 0; r < 8; ++r) {
            float a0 = sc[0].f[r] * scale;
            float a1 = sc[1].f[r] * scale;
            float mx = fmaxf(a0, a1);
            #pragma unroll
            for (int off = 1; off < 16; off <<= 1)
                mx = fmaxf(mx, __shfl_xor(mx, off, 16));
            const float mnew = fmaxf(m[r], mx);
            const float corr = __expf(m[r] - mnew);
            p0[r] = __expf(a0 - mnew);
            p1[r] = __expf(a1 - mnew);
            float rs = p0[r] + p1[r];
            #pragma unroll
            for (int off = 1; off < 16; off <<= 1)
                rs += __shfl_xor(rs, off, 16);
            l[r] = l[r] * corr + rs;
            m[r] = mnew;
            #pragma unroll
            for (int f = 0; f < 4; ++f) o[f].f[r] *= corr;
        }

        // ---- P: C layout -> A layout via per-wave LDS ---------------------
        #pragma unroll
        for (int r = 0; r < 8; ++r) {
            pbuf[(r + 8 * half) * 32 + ln]      = (__bf16)p0[r];
            pbuf[(r + 8 * half) * 32 + 16 + ln] = (__bf16)p1[r];
        }
        asm volatile("s_wait_dscnt 0" ::: "memory");  // wave-local DS ordering
        AFrag pa;
        pa.h[0] = *(const v8bf*)(pbuf + ln * 32 + 16 * half);
        pa.h[1] = *(const v8bf*)(pbuf + ln * 32 + 16 * half + 8);

        // ---- O += P @ V  (V frags from LDS V^T chunk) ---------------------
        const __bf16* vl = &vbuf[cur][0];
        #pragma unroll
        for (int f = 0; f < 4; ++f) {
            const __bf16* vrow = vl + (f * 16 + ln) * 32 + 16 * half;
            BFrag vb;
            vb.h[0] = *(const v8bf*)(vrow);
            vb.h[1] = *(const v8bf*)(vrow + 8);
            o[f].v = __builtin_amdgcn_wmma_f32_16x16x32_bf16(
                false, pa.v, false, vb.v, (short)0, o[f].v, false, false);
        }

        if (wave == 0) __builtin_amdgcn_s_wait_tensorcnt(0);  // next chunk landed
        __syncthreads();                                      // publish to WG
        cur = nxt;
    }

    // ---- epilogue: O /= l, merge heads -> [B,S,H] fp32 --------------------
    const int b    = bh >> 4;
    const int head = bh & 15;
    #pragma unroll
    for (int r = 0; r < 8; ++r) {
        const float inv  = 1.0f / l[r];
        const int   srow = s0 + r + 8 * half;
        float* orow = out + ((size_t)b * S_ + srow) * H_ + head * HD_;
        #pragma unroll
        for (int f = 0; f < 4; ++f)
            orow[f * 16 + ln] = o[f].f[r] * inv;
    }
}

extern "C" void kernel_launch(void* const* d_in, const int* in_sizes, int n_in,
                              void* d_out, int out_size, void* d_ws, size_t ws_size,
                              hipStream_t stream) {
    (void)in_sizes; (void)n_in; (void)out_size; (void)ws_size;
    const float* hs = (const float*)d_in[0];
    const float* Wq = (const float*)d_in[1];
    const float* bq = (const float*)d_in[2];
    const float* Wk = (const float*)d_in[3];
    const float* bk = (const float*)d_in[4];
    const float* Wv = (const float*)d_in[5];
    const float* bv = (const float*)d_in[6];
    float* out = (float*)d_out;

    const size_t QSZ = (size_t)B_ * NH_ * S_ * HD_;   // 8Mi elements
    const size_t WSZ = (size_t)H_ * H_;               // 1Mi elements
    __bf16* Qb  = (__bf16*)d_ws;
    __bf16* Kb  = Qb  + QSZ;
    __bf16* VTb = Kb  + QSZ;
    __bf16* hsb = VTb + QSZ;
    __bf16* Wqb = hsb + QSZ;
    __bf16* Wkb = Wqb + WSZ;
    __bf16* Wvb = Wkb + WSZ;                          // ~70MB workspace total

    // K0: fp32 -> bf16 (hs: 8Mi floats, each W: 1Mi floats), 8 elems/thread
    hipLaunchKernelGGL(cvt_bf16_kernel, dim3(4096), dim3(256), 0, stream,
                       hs, hsb, (int)(QSZ / 8));
    hipLaunchKernelGGL(cvt_bf16_kernel, dim3(512), dim3(256), 0, stream,
                       Wq, Wqb, (int)(WSZ / 8));
    hipLaunchKernelGGL(cvt_bf16_kernel, dim3(512), dim3(256), 0, stream,
                       Wk, Wkb, (int)(WSZ / 8));
    hipLaunchKernelGGL(cvt_bf16_kernel, dim3(512), dim3(256), 0, stream,
                       Wv, Wvb, (int)(WSZ / 8));
    // K1: 3 mats * 512 row tiles * 8 strips = 12288 waves / 4 per block
    hipLaunchKernelGGL(qkv_proj_kernel, dim3(3072), dim3(128), 0, stream,
                       hsb, Wqb, Wkb, Wvb, bq, bk, bv, Qb, Kb, VTb);
    // K2: 128 bh * 16 row groups = 2048 workgroups of 4 waves
    hipLaunchKernelGGL(flash_attn_kernel, dim3(2048), dim3(128), 0, stream,
                       Qb, Kb, VTb, out);
}
